// Kenn_13417477833078
// MI455X (gfx1250) — compile-verified, matrix operand order checked by reading, not run on hardware
//
#include <hip/hip_runtime.h>
#include <hip/hip_bf16.h>

// ---- problem constants (match reference) ----
#define N_NODES   20000
#define N_EDGES   640000
#define N_FEATS   3703
#define HIDDEN    50
#define N_CLASSES 6

typedef __attribute__((ext_vector_type(16))) __bf16 v16bf;
typedef __attribute__((ext_vector_type(8)))  float  v8f;

union Frag16 { v16bf v; __bf16 h[16]; };

// ---------------------------------------------------------------------------
// Pre-pack: W[K,N] fp32 -> bf16 WMMA-B fragment buffer, laid out as
// [chunk][nt][lane 0..31][slot 0..15] so each lane's 16 operands are one
// contiguous, 32B-aligned v16bf. Zero-padded at K/N edges -> the GEMM hot
// loop carries no guards at all.
// Fragment map (mirror of A): lane<16 = col, K{0..7,16..23}; lane>=16 -> +8.
// ---------------------------------------------------------------------------
template <int NT>
__global__ __launch_bounds__(256) void pack_w_frag(
    const float* __restrict__ W, __bf16* __restrict__ frag,
    int K, int N, int nCh)
{
    int t = blockIdx.x * 256 + threadIdx.x;
    int total = nCh * NT * 512;
    if (t >= total) return;
    int slot = t & 15;
    int lane = (t >> 4) & 31;
    int nt   = (t >> 9) % NT;
    int ch   = t / (NT * 512);
    int col  = lane & 15;
    int hi   = lane >> 4;
    int k    = (slot & 7) + hi * 8 + ((slot & 8) ? 16 : 0);
    int kg   = ch * 32 + k;
    int n    = nt * 16 + col;
    float w  = (kg < K && n < N) ? W[kg * N + n] : 0.0f;
    frag[t] = (__bf16)w;
}

// ---------------------------------------------------------------------------
// C = act(A[M,K] @ W[K,N] + b), W pre-packed as bf16 fragments.
// Block: 256 threads = 8 waves (wave32); wave w owns rows [128*bid+16w, +16).
// No LDS, no barriers: A streamed from HBM (the roofline term), packed W
// streamed from L2, K in chunks of 32.
// ---------------------------------------------------------------------------
template <int NT, bool RELU>
__global__ __launch_bounds__(256) void gemm_bias_act(
    const float* __restrict__ A, const __bf16* __restrict__ Wfrag,
    const float* __restrict__ bias, float* __restrict__ out,
    int M, int K, int N)
{
    const int tid    = threadIdx.x;
    const int lane   = tid & 31;
    const int wave   = tid >> 5;
    const int laneLo = lane & 15;
    const int laneHi = lane >> 4;
    const int rowBase = blockIdx.x * 128 + wave * 16;

    // Clamp instead of guard: an OOB A-row only feeds D-rows >= M, which are
    // never stored, so clamped unguarded loads are safe and branch-free.
    int rowC = rowBase + laneLo;
    if (rowC > M - 1) rowC = M - 1;
    const float* __restrict__ Arow = A + (size_t)rowC * K;
    const __bf16* __restrict__ bp  = Wfrag + (size_t)lane * 16;

    v8f acc[NT] = {};
    const int kMain = K & ~31;   // full 32-wide chunks, guard-free

    int ch = 0;
    for (int kc = 0; kc < kMain; kc += 32, ++ch) {
        const float* __restrict__ ap = Arow + kc + laneHi * 8;
        Frag16 a;
        #pragma unroll
        for (int j = 0; j < 8; ++j) {
            a.h[j]     = (__bf16)ap[j];        // lane<16: K{0..7},  lane>=16: K{8..15}
            a.h[j + 8] = (__bf16)ap[j + 16];   // lane<16: K{16..23},lane>=16: K{24..31}
        }
        if (kc + 32 < K) __builtin_prefetch(ap + 32, 0, 3);  // global_prefetch_b8

        #pragma unroll
        for (int nt = 0; nt < NT; ++nt) {
            v16bf b = *reinterpret_cast<const v16bf*>(bp + (size_t)(ch * NT + nt) * 512);
            acc[nt] = __builtin_amdgcn_wmma_f32_16x16x32_bf16(
                false, a.v, false, b, (short)0, acc[nt], false, false);
        }
    }

    // K tail chunk (guarded loads); packed W already zero-padded
    if (kMain < K) {
        Frag16 a;
        #pragma unroll
        for (int j = 0; j < 8; ++j) {
            int k0 = kMain + laneHi * 8 + j;
            int k1 = k0 + 16;
            a.h[j]     = (__bf16)(k0 < K ? Arow[k0] : 0.0f);
            a.h[j + 8] = (__bf16)(k1 < K ? Arow[k1] : 0.0f);
        }
        #pragma unroll
        for (int nt = 0; nt < NT; ++nt) {
            v16bf b = *reinterpret_cast<const v16bf*>(bp + (size_t)(ch * NT + nt) * 512);
            acc[nt] = __builtin_amdgcn_wmma_f32_16x16x32_bf16(
                false, a.v, false, b, (short)0, acc[nt], false, false);
        }
    }

    // epilogue: bias + relu + store (C/D layout: VGPR v -> row v + 8*laneHi)
    #pragma unroll
    for (int nt = 0; nt < NT; ++nt) {
        int col = nt * 16 + laneLo;
        if (col >= N) continue;
        float bv = bias[col];
        #pragma unroll
        for (int v = 0; v < 8; ++v) {
            int r = rowBase + v + laneHi * 8;
            if (r < M) {
                float val = acc[nt][v] + bv;
                if (RELU) val = fmaxf(val, 0.0f);
                out[(size_t)r * N + col] = val;
            }
        }
    }
}

// ---------------------------------------------------------------------------
// KENN edge kernel: per-edge 3-way softmax per class, atomic scatter of deltas
// ---------------------------------------------------------------------------
__global__ __launch_bounds__(256) void kenn_edge(
    const float* __restrict__ z, const float* __restrict__ rel,
    const int* __restrict__ sx, const int* __restrict__ sy,
    const float* __restrict__ cw, float* __restrict__ dz)
{
    int e = blockIdx.x * 256 + threadIdx.x;
    if (e >= N_EDGES) return;
    int ix = sx[e];
    int iy = sy[e];
    float r = rel[e];
    #pragma unroll
    for (int c = 0; c < N_CLASSES; ++c) {
        float v0 = -z[ix * N_CLASSES + c];   // -C(x)
        float v1 =  z[iy * N_CLASSES + c];   //  C(y)
        float v2 = -r;                       // -Cite(x,y)
        float m  = fmaxf(fmaxf(v0, v1), v2);
        float e0 = __expf(v0 - m);
        float e1 = __expf(v1 - m);
        float e2 = __expf(v2 - m);
        float inv = 1.0f / (e0 + e1 + e2);
        float w  = cw[c];
        atomicAdd(&dz[ix * N_CLASSES + c], -w * e0 * inv);
        atomicAdd(&dz[iy * N_CLASSES + c],  w * e1 * inv);
    }
}

__global__ __launch_bounds__(256) void zero_kernel(float* __restrict__ p, int n) {
    int i = blockIdx.x * 256 + threadIdx.x;
    if (i < n) p[i] = 0.0f;
}

// z += dz; dz = 0   (prepares dz for the next KENN layer)
__global__ __launch_bounds__(256) void add_clear_kernel(float* __restrict__ z,
                                                        float* __restrict__ dz, int n) {
    int i = blockIdx.x * 256 + threadIdx.x;
    if (i < n) { z[i] += dz[i]; dz[i] = 0.0f; }
}

// out = softmax(z + dz) per node (fuses the last KENN apply)
__global__ __launch_bounds__(256) void softmax_fused_kernel(
    const float* __restrict__ z, const float* __restrict__ dz,
    float* __restrict__ out)
{
    int i = blockIdx.x * 256 + threadIdx.x;
    if (i >= N_NODES) return;
    float v[N_CLASSES];
    float m = -1e30f;
    #pragma unroll
    for (int c = 0; c < N_CLASSES; ++c) {
        v[c] = z[i * N_CLASSES + c] + dz[i * N_CLASSES + c];
        m = fmaxf(m, v[c]);
    }
    float s = 0.0f;
    #pragma unroll
    for (int c = 0; c < N_CLASSES; ++c) { v[c] = __expf(v[c] - m); s += v[c]; }
    float inv = 1.0f / s;
    #pragma unroll
    for (int c = 0; c < N_CLASSES; ++c) out[i * N_CLASSES + c] = v[c] * inv;
}

// ---------------------------------------------------------------------------
extern "C" void kernel_launch(void* const* d_in, const int* in_sizes, int n_in,
                              void* d_out, int out_size, void* d_ws, size_t ws_size,
                              hipStream_t stream)
{
    (void)in_sizes; (void)n_in; (void)out_size; (void)ws_size;
    const float* features  = (const float*)d_in[0];   // [20000, 3703]
    const float* relations = (const float*)d_in[1];   // [640000, 1]
    const int*   sx        = (const int*)  d_in[2];   // [640000]
    const int*   sy        = (const int*)  d_in[3];   // [640000]
    const float* W1 = (const float*)d_in[4];  const float* b1 = (const float*)d_in[5];
    const float* W2 = (const float*)d_in[6];  const float* b2 = (const float*)d_in[7];
    const float* W3 = (const float*)d_in[8];  const float* b3 = (const float*)d_in[9];
    const float* W4 = (const float*)d_in[10]; const float* b4 = (const float*)d_in[11];
    const float* cw = (const float*)d_in[12]; // [3, 6]

    // chunk counts
    const int ch1 = (N_FEATS + 31) / 32;   // 116
    const int ch2 = (HIDDEN + 31) / 32;    // 2

    // workspace layout (floats first, then 32B-aligned bf16 fragment buffers)
    float* x1 = (float*)d_ws;                                 // [20000, 50]
    float* x2 = x1 + (size_t)N_NODES * HIDDEN;                // [20000, 50]
    float* z  = x2 + (size_t)N_NODES * HIDDEN;                // [20000, 6]
    float* dz = z  + (size_t)N_NODES * N_CLASSES;             // [20000, 6]
    __bf16* f1 = (__bf16*)(dz + (size_t)N_NODES * N_CLASSES); // base is 64B-aligned
    __bf16* f2 = f1 + (size_t)ch1 * 4 * 512;                  // 237568 elems
    __bf16* f3 = f2 + (size_t)ch2 * 4 * 512;                  // 4096
    __bf16* f4 = f3 + (size_t)ch2 * 4 * 512;                  // 4096 (+1024 for f4)

    dim3 blk(256);
    int mBlocks = (N_NODES + 127) / 128;

    // pre-pack weights into bf16 fragment layout (tiny one-off kernels)
    pack_w_frag<4><<<(ch1 * 4 * 512 + 255) / 256, blk, 0, stream>>>(W1, f1, N_FEATS, HIDDEN, ch1);
    pack_w_frag<4><<<(ch2 * 4 * 512 + 255) / 256, blk, 0, stream>>>(W2, f2, HIDDEN, HIDDEN, ch2);
    pack_w_frag<4><<<(ch2 * 4 * 512 + 255) / 256, blk, 0, stream>>>(W3, f3, HIDDEN, HIDDEN, ch2);
    pack_w_frag<1><<<(ch2 * 1 * 512 + 255) / 256, blk, 0, stream>>>(W4, f4, HIDDEN, N_CLASSES, ch2);

    // MLP (bf16 WMMA, f32 accumulate). x3 reuses x1's buffer.
    gemm_bias_act<4, true ><<<mBlocks, blk, 0, stream>>>(features, f1, b1, x1, N_NODES, N_FEATS, HIDDEN);
    gemm_bias_act<4, true ><<<mBlocks, blk, 0, stream>>>(x1, f2, b2, x2, N_NODES, HIDDEN, HIDDEN);
    gemm_bias_act<4, true ><<<mBlocks, blk, 0, stream>>>(x2, f3, b3, x1, N_NODES, HIDDEN, HIDDEN);
    gemm_bias_act<1, false><<<mBlocks, blk, 0, stream>>>(x1, f4, b4, z,  N_NODES, HIDDEN, N_CLASSES);

    // 3 KENN layers; dz cleared by add_clear for subsequent layers
    int zElems  = N_NODES * N_CLASSES;
    int zBlocks = (zElems + 255) / 256;
    int eBlocks = (N_EDGES + 255) / 256;

    zero_kernel<<<zBlocks, blk, 0, stream>>>(dz, zElems);
    kenn_edge<<<eBlocks, blk, 0, stream>>>(z, relations, sx, sy, cw + 0 * N_CLASSES, dz);
    add_clear_kernel<<<zBlocks, blk, 0, stream>>>(z, dz, zElems);
    kenn_edge<<<eBlocks, blk, 0, stream>>>(z, relations, sx, sy, cw + 1 * N_CLASSES, dz);
    add_clear_kernel<<<zBlocks, blk, 0, stream>>>(z, dz, zElems);
    kenn_edge<<<eBlocks, blk, 0, stream>>>(z, relations, sx, sy, cw + 2 * N_CLASSES, dz);

    // out = softmax(z + dz)  (fused final apply)
    softmax_fused_kernel<<<(N_NODES + 255) / 256, blk, 0, stream>>>(z, dz, (float*)d_out);
}
